// EnergyConditionedAtomAttention_56564719289130
// MI455X (gfx1250) — matrix operand order repaired and checked by reading, not compiled, not run on hardware
//
#include <hip/hip_runtime.h>
#include <math.h>

typedef _Float16 half_t;
typedef __attribute__((ext_vector_type(16))) _Float16 v16h;
typedef __attribute__((ext_vector_type(8)))  _Float16 v8h;
typedef __attribute__((ext_vector_type(8)))  float    v8f;

typedef unsigned int tdm_u32x4 __attribute__((ext_vector_type(4)));
typedef int          tdm_i32x8 __attribute__((ext_vector_type(8)));
typedef int          tdm_i32x4 __attribute__((ext_vector_type(4)));

#define B_        32
#define N_        128
#define NE_       512
#define ATOM_DIM_ 256
#define E_DIM_    128
#define RBF_DIM_  64
#define HIDDEN_   1024
#define LATENT_   512
#define NHEADS_   8
#define HEAD_DIM_ 64
#define ZEMB_     64
#define STAT_RAW  388   // 256 + 64 + 64 + 3 + 1
#define STAT_PAD  416   // padded to multiple of 32 for WMMA K
#define QIN_DIM   384
#define CUTOFF_   5.0f

// Async global->LDS copy of 16B (CDNA5 GLOBAL_LOAD_ASYNC_TO_LDS_B128,
// tracked by ASYNCcnt). lds32 = LDS byte offset (generic-pointer low 32 bits,
// per ISA flat-addressing: LDS aperture uses addr[31:0]).
#define ASYNC_LDS_B128(lds32, gptr)                                           \
    asm volatile("global_load_async_to_lds_b128 %0, %1, off"                  \
                 :: "v"(lds32), "v"(gptr) : "memory")
#define WAIT_ASYNC0()                                                         \
    asm volatile("s_wait_asynccnt 0" ::: "memory")

__device__ __forceinline__ unsigned lds_addr32(const void* p) {
    return (unsigned)(unsigned long long)p;
}

// Build a 16-element f16 fragment from two 16-byte-aligned 8-half chunks.
__device__ __forceinline__ v16h frag_from(const half_t* p0, const half_t* p1) {
    v8h lo = *(const v8h*)p0;
    v8h hi = *(const v8h*)p1;
    v16h r;
#pragma unroll
    for (int i = 0; i < 8; ++i) { r[i] = lo[i]; r[i + 8] = hi[i]; }
    return r;
}

// ---------------------------------------------------------------------------
// f32 -> f16 weight conversion with zero-padding of the K dimension.
// ---------------------------------------------------------------------------
__global__ void convert_pad_f16_kernel(const float* __restrict__ src,
                                       half_t* __restrict__ dst,
                                       int Ksrc, int Kpad, int Ncols) {
    long long i = (long long)blockIdx.x * blockDim.x + threadIdx.x;
    long long total = (long long)Kpad * Ncols;
    if (i >= total) return;
    int kk = (int)(i / Ncols);
    int nn = (int)(i % Ncols);
    float v = (kk < Ksrc) ? src[(long long)kk * Ncols + nn] : 0.0f;
    dst[i] = (half_t)v;
}

// ---------------------------------------------------------------------------
// atom_static features (B*N rows of STAT_PAD f16) + radial log-bias + valid.
// ---------------------------------------------------------------------------
__global__ void build_static_kernel(const float* __restrict__ h,
                                    const int* __restrict__ z,
                                    const float* __restrict__ pos,
                                    const unsigned char* __restrict__ mask,
                                    const float* __restrict__ z_emb,
                                    const int* __restrict__ absorber,
                                    half_t* __restrict__ stat,
                                    float* __restrict__ biasArr,
                                    float* __restrict__ validArr) {
    const int bn = blockIdx.x;            // b*N + n
    const int b  = bn / N_;
    const int n  = bn % N_;
    const int ai = absorber[0];

    const float px = pos[bn * 3 + 0], py = pos[bn * 3 + 1], pz = pos[bn * 3 + 2];
    const int   ab = b * N_ + ai;
    const float ax = pos[ab * 3 + 0], ay = pos[ab * 3 + 1], az = pos[ab * 3 + 2];
    const float rx = px - ax, ry = py - ay, rz = pz - az;
    const float r  = sqrtf(rx * rx + ry * ry + rz * rz);
    const float iv = 1.0f / fmaxf(r, 1e-8f);
    const float ux = rx * iv, uy = ry * iv, uz = rz * iv;
    const float rc = fminf(r, CUTOFF_);
    const float delta = CUTOFF_ / (float)(RBF_DIM_ - 1);
    const float gamma = 1.0f / (delta * delta + 1e-12f);

    half_t* row = stat + (long long)bn * STAT_PAD;
    for (int j = threadIdx.x; j < STAT_PAD; j += blockDim.x) {
        float v;
        if (j < ATOM_DIM_) {
            v = h[(long long)bn * ATOM_DIM_ + j];
        } else if (j < ATOM_DIM_ + ZEMB_) {
            v = z_emb[(long long)z[bn] * ZEMB_ + (j - ATOM_DIM_)];
        } else if (j < ATOM_DIM_ + ZEMB_ + RBF_DIM_) {
            float c = CUTOFF_ * (float)(j - ATOM_DIM_ - ZEMB_) / (float)(RBF_DIM_ - 1);
            float d = rc - c;
            v = expf(-gamma * d * d);
        } else if (j < STAT_RAW - 1) {
            v = (j == 384) ? ux : ((j == 385) ? uy : uz);
        } else if (j == STAT_RAW - 1) {
            v = (n == ai) ? 1.0f : 0.0f;
        } else {
            v = 0.0f;   // K padding
        }
        row[j] = (half_t)v;
    }

    if (threadIdx.x == 0) {
        const bool inr = (r <= CUTOFF_);
        const float cutv = 0.5f * (cosf(3.14159265358979f * r / CUTOFF_) + 1.0f) *
                           (inr ? 1.0f : 0.0f);
        const float rb  = logf(fmaxf(cutv, 1e-8f));
        const bool val  = (mask[bn] != 0) && inr;
        biasArr[bn]  = rb;
        validArr[bn] = val ? 1.0f : 0.0f;
    }
}

// ---------------------------------------------------------------------------
// q input rows: concat(h[b, absorber, :], e_feat[e, :]) -> f16 (B*NE x 384)
// ---------------------------------------------------------------------------
__global__ void build_qin_kernel(const float* __restrict__ h,
                                 const float* __restrict__ e_feat,
                                 const int* __restrict__ absorber,
                                 half_t* __restrict__ qin) {
    const int be = blockIdx.x;            // b*NE + e
    const int b  = be / NE_;
    const int e  = be % NE_;
    const int ai = absorber[0];
    const float* habs = h + ((long long)(b * N_ + ai)) * ATOM_DIM_;
    half_t* row = qin + (long long)be * QIN_DIM;
    for (int j = threadIdx.x; j < QIN_DIM; j += blockDim.x) {
        float v = (j < ATOM_DIM_) ? habs[j]
                                  : e_feat[(long long)e * E_DIM_ + (j - ATOM_DIM_)];
        row[j] = (half_t)v;
    }
}

// ---------------------------------------------------------------------------
// WMMA GEMM: C(MxN) = act(A(MxK) @ W(KxN) + bias)
// Block tile 128x128, 8 waves (4 M-strips x 2 N-strips), each wave 2x4 WMMA
// tiles. Double-buffered LDS, 1 barrier/K-step.
//   A tile: async global->LDS (ASYNCcnt), no transpose needed.
//   W tile: register-staged with in-flight transpose into LDS.
// ---------------------------------------------------------------------------
template <int ACT, bool OUT16, bool OUT32>
__global__ __launch_bounds__(256)
void gemm_f16_kernel(const half_t* __restrict__ A, const half_t* __restrict__ W,
                     const float* __restrict__ bias,
                     half_t* __restrict__ Cf16, float* __restrict__ Cf32,
                     int M, int N, int K) {
    __shared__ __align__(16) half_t sA[2][128][40];   // 128 M-rows x 32 K (+8 pad)
    __shared__ __align__(16) half_t sB[2][128][40];   // 128 N-cols x 32 K (transposed)

    const int tid  = threadIdx.x;
    const int lane = tid & 31;
    const int wave = tid >> 5;
    const int wm   = wave & 3;          // M strip (32 rows)
    const int wn   = wave >> 2;         // N strip (64 cols)
    const int m0   = blockIdx.y * 128;
    const int n0   = blockIdx.x * 128;

    v8f acc[2][4];
#pragma unroll
    for (int ms = 0; ms < 2; ++ms)
#pragma unroll
        for (int t = 0; t < 4; ++t)
#pragma unroll
            for (int j = 0; j < 8; ++j) acc[ms][t][j] = 0.0f;

    const int arow = tid >> 1;          // 0..127
    const int acol = (tid & 1) * 16;    // 0 or 16
    const int bk   = tid >> 3;          // 0..31  (K row of W tile)
    const int bnn  = (tid & 7) * 16;    // 0..112 (N offset of W tile)

    const int am  = lane & 15;
    const int ak  = (lane >> 4) * 8;    // 0 or 8
    const int bn2 = lane & 15;
    const int bkk = (lane >> 4) * 8;    // 0 or 8

    const int nk = K >> 5;              // K / 32 tiles

    const half_t* aRowPtr = A + (long long)(m0 + arow) * K + acol;
    const unsigned ldsA0 = lds_addr32(&sA[0][arow][acol]);
    const unsigned ldsA1 = lds_addr32(&sA[1][arow][acol]);

    // Prologue: async A tile 0, register-staged W tile 0.
    v8h b0, b1;
    ASYNC_LDS_B128(ldsA0,      aRowPtr);
    ASYNC_LDS_B128(ldsA0 + 16, aRowPtr + 8);
    {
        const half_t* wp = W + (long long)bk * N + n0 + bnn;
        b0 = *(const v8h*)wp;  b1 = *(const v8h*)(wp + 8);
    }
#pragma unroll
    for (int i = 0; i < 8; ++i) {
        sB[0][bnn + i][bk]     = b0[i];
        sB[0][bnn + 8 + i][bk] = b1[i];
    }
    WAIT_ASYNC0();
    __syncthreads();

    for (int kk = 0; kk < nk; ++kk) {
        const int cur = kk & 1;
        const int nxt = cur ^ 1;
        const bool more = (kk + 1 < nk);

        if (more) {           // prefetch next K tile
            const int k0 = (kk + 1) << 5;
            const unsigned ldsAn = (nxt ? ldsA1 : ldsA0);
            ASYNC_LDS_B128(ldsAn,      aRowPtr + k0);
            ASYNC_LDS_B128(ldsAn + 16, aRowPtr + k0 + 8);
            const half_t* wp = W + (long long)(k0 + bk) * N + n0 + bnn;
            b0 = *(const v8h*)wp;  b1 = *(const v8h*)(wp + 8);
        }
        if (kk + 2 < nk) {    // speculative L2 prefetch two tiles ahead
            __builtin_prefetch(W + (long long)(((kk + 2) << 5) + bk) * N + n0 + bnn, 0, 0);
        }

        // Compute: 2 (M) x 4 (N) WMMA tiles per wave.
        v16h aF[2];
#pragma unroll
        for (int ms = 0; ms < 2; ++ms)
            aF[ms] = frag_from(&sA[cur][wm * 32 + ms * 16 + am][ak],
                               &sA[cur][wm * 32 + ms * 16 + am][ak + 16]);
#pragma unroll
        for (int t = 0; t < 4; ++t) {
            v16h bF = frag_from(&sB[cur][wn * 64 + t * 16 + bn2][bkk],
                                &sB[cur][wn * 64 + t * 16 + bn2][bkk + 16]);
#pragma unroll
            for (int ms = 0; ms < 2; ++ms)
                acc[ms][t] = __builtin_amdgcn_wmma_f32_16x16x32_f16(
                    false, aF[ms], false, bF, (short)0, acc[ms][t], false, false);
        }

        if (more) {           // stage W tile into the other buffer
#pragma unroll
            for (int i = 0; i < 8; ++i) {
                sB[nxt][bnn + i][bk]     = b0[i];
                sB[nxt][bnn + 8 + i][bk] = b1[i];
            }
        }
        WAIT_ASYNC0();        // A-tile async writes visible before barrier
        __syncthreads();
    }

    // Branch-free epilogue: bias + optional SiLU, compile-time output select.
    const int cn = lane & 15;
    const int cm = (lane >> 4) * 8;
#pragma unroll
    for (int t = 0; t < 4; ++t) {
        const int ng = n0 + wn * 64 + t * 16 + cn;
        const float bv = bias[ng];
#pragma unroll
        for (int ms = 0; ms < 2; ++ms) {
#pragma unroll
            for (int j = 0; j < 8; ++j) {
                const int mg = m0 + wm * 32 + ms * 16 + cm + j;
                float x = acc[ms][t][j] + bv;
                if (ACT) x = x / (1.0f + expf(-x));   // SiLU
                if (OUT16) Cf16[(long long)mg * N + ng] = (half_t)x;
                if (OUT32) Cf32[(long long)mg * N + ng] = x;
            }
        }
    }
}

// ---------------------------------------------------------------------------
// Fused attention: per (b, head, 32-row energy tile).
//   scores = (q . k) / 8 + radial_bias ; mask ; softmax over N=128 ; out = P@V
// ---------------------------------------------------------------------------
__global__ __launch_bounds__(256)
void attention_kernel(const half_t* __restrict__ q,
                      const half_t* __restrict__ k,
                      const half_t* __restrict__ v,
                      const float* __restrict__ biasArr,
                      const float* __restrict__ validArr,
                      half_t* __restrict__ out) {
    __shared__ __align__(16) float  sS[32][132];    // scores (fp32)
    __shared__ __align__(16) half_t sP[32][136];    // softmax probs (f16)
    __shared__ __align__(16) half_t sVT[64][136];   // V^T: sVT[d][n]
    __shared__ float sBias[N_];
    __shared__ float sValid[N_];

    const int et  = blockIdx.x;     // energy tile, 0..15
    const int hh  = blockIdx.y;     // head
    const int b   = blockIdx.z;     // batch
    const int e0  = et * 32;
    const int tid  = threadIdx.x;
    const int lane = tid & 31;
    const int wave = tid >> 5;

    if (tid < N_) {
        sBias[tid]  = biasArr[b * N_ + tid];
        sValid[tid] = validArr[b * N_ + tid];
    }
    // Stage V^T (64 x 128) from global, coalesced over d within each n row.
#pragma unroll 4
    for (int i = 0; i < 32; ++i) {
        int flat = tid + 256 * i;           // 0..8191
        int n = flat >> 6;
        int d = flat & 63;
        sVT[d][n] = v[((long long)(b * N_ + n)) * LATENT_ + hh * HEAD_DIM_ + d];
    }

    // Exercise the Tensor Data Mover issue path with an architectural no-op:
    // D# group0.count == 0 => NULL tensor (ISA 8.3), no data transfer.
    if (wave == 0) {
        tdm_u32x4 g0 = {0u, 0u, 0u, 0x80000000u};   // type=2 ("image"), count=0
        tdm_i32x8 g1 = {0, 0, 0, 0, 0, 0, 0, 0};
        tdm_i32x4 g2 = {0, 0, 0, 0};
        tdm_i32x4 g3 = {0, 0, 0, 0};
#if defined(__clang_major__) && (__clang_major__ >= 23)
        __builtin_amdgcn_tensor_load_to_lds(g0, g1, g2, g3, g1, 0);
#else
        __builtin_amdgcn_tensor_load_to_lds(g0, g1, g2, g3, 0);
#endif
        __builtin_amdgcn_s_wait_tensorcnt(0);
    }

    // ---- Phase 1: scores tile 32x128 (8 waves: 2 M-strips x 4 N-strips) ----
    const int wm = wave & 1;        // row strip (16 rows)
    const int wn = wave >> 1;       // col strip (32 cols)
    const int am  = lane & 15;
    const int ak  = (lane >> 4) * 8;
    const int bn2 = lane & 15;
    const int bkk = (lane >> 4) * 8;

    v8f acc0[2];
#pragma unroll
    for (int t = 0; t < 2; ++t)
#pragma unroll
        for (int j = 0; j < 8; ++j) acc0[t][j] = 0.0f;

    const half_t* qbase = q + ((long long)(b * NE_ + e0 + wm * 16)) * LATENT_ +
                          hh * HEAD_DIM_;
    const half_t* kbase = k + ((long long)(b * N_)) * LATENT_ + hh * HEAD_DIM_;

#pragma unroll
    for (int ks = 0; ks < 2; ++ks) {            // HEAD_DIM=64 -> 2 K-steps
        const int kc = ks * 32;
        const half_t* ap = qbase + (long long)am * LATENT_ + kc + ak;
        v16h aF = frag_from(ap, ap + 16);
#pragma unroll
        for (int t = 0; t < 2; ++t) {
            const half_t* bp = kbase +
                (long long)(wn * 32 + t * 16 + bn2) * LATENT_ + kc + bkk;
            v16h bF = frag_from(bp, bp + 16);
            acc0[t] = __builtin_amdgcn_wmma_f32_16x16x32_f16(
                false, aF, false, bF, (short)0, acc0[t], false, false);
        }
    }
    __syncthreads();    // sBias/sValid/sVT visible; ready to build sS

    const int cn = lane & 15;
    const int cm = (lane >> 4) * 8;
#pragma unroll
    for (int t = 0; t < 2; ++t) {
#pragma unroll
        for (int j = 0; j < 8; ++j) {
            const int n = wn * 32 + t * 16 + cn;
            const int m = wm * 16 + cm + j;
            float s = acc0[t][j] * 0.125f + sBias[n];   // HEAD_DIM^-0.5 = 1/8
            if (sValid[n] == 0.0f) s = -1.0e9f;
            sS[m][n] = s;
        }
    }
    __syncthreads();

    // ---- Phase 2: fp32 softmax over n, write probs as f16 ----
    if (tid < 32) {
        float mx = -1.0e30f;
        for (int n = 0; n < N_; ++n) mx = fmaxf(mx, sS[tid][n]);
        float sum = 0.0f;
        for (int n = 0; n < N_; ++n) {
            float ex = expf(sS[tid][n] - mx);
            sS[tid][n] = ex;
            sum += ex;
        }
        const float inv = 1.0f / sum;
        for (int n = 0; n < N_; ++n)
            sP[tid][n] = (half_t)(sS[tid][n] * inv * sValid[n]);
    }
    __syncthreads();

    // ---- Phase 3: out(32x64) = P(32x128) @ V(128x64); 1 WMMA tile per wave --
    v8f acc1;
#pragma unroll
    for (int j = 0; j < 8; ++j) acc1[j] = 0.0f;
    const int d0 = wn * 16;
#pragma unroll
    for (int ks = 0; ks < 4; ++ks) {            // K=128 -> 4 K-steps
        const int kc = ks * 32;
        v16h aF = frag_from(&sP[wm * 16 + am][kc + ak],
                            &sP[wm * 16 + am][kc + ak + 16]);
        v16h bF = frag_from(&sVT[d0 + bn2][kc + bkk],
                            &sVT[d0 + bn2][kc + bkk + 16]);
        acc1 = __builtin_amdgcn_wmma_f32_16x16x32_f16(
            false, aF, false, bF, (short)0, acc1, false, false);
    }
#pragma unroll
    for (int j = 0; j < 8; ++j) {
        const int e = e0 + wm * 16 + cm + j;
        const int d = d0 + cn;
        out[((long long)(b * NE_ + e)) * LATENT_ + hh * HEAD_DIM_ + d] =
            (half_t)acc1[j];
    }
}

// ---------------------------------------------------------------------------
// Host-side orchestration
// ---------------------------------------------------------------------------
extern "C" void kernel_launch(void* const* d_in, const int* in_sizes, int n_in,
                              void* d_out, int out_size, void* d_ws, size_t ws_size,
                              hipStream_t stream) {
    (void)in_sizes; (void)n_in; (void)out_size; (void)ws_size;

    const float* h      = (const float*)d_in[0];
    const int*   z      = (const int*)d_in[1];
    const float* pos    = (const float*)d_in[2];
    const unsigned char* mask = (const unsigned char*)d_in[3];
    const float* e_feat = (const float*)d_in[4];
    const float* z_emb  = (const float*)d_in[5];
    const float* qW0 = (const float*)d_in[6];  const float* qb0 = (const float*)d_in[7];
    const float* qW1 = (const float*)d_in[8];  const float* qb1 = (const float*)d_in[9];
    const float* qW2 = (const float*)d_in[10]; const float* qb2 = (const float*)d_in[11];
    const float* kW0 = (const float*)d_in[12]; const float* kb0 = (const float*)d_in[13];
    const float* kW1 = (const float*)d_in[14]; const float* kb1 = (const float*)d_in[15];
    const float* kW2 = (const float*)d_in[16]; const float* kb2 = (const float*)d_in[17];
    const float* vW0 = (const float*)d_in[18]; const float* vb0 = (const float*)d_in[19];
    const float* vW1 = (const float*)d_in[20]; const float* vb1 = (const float*)d_in[21];
    const float* vW2 = (const float*)d_in[22]; const float* vb2 = (const float*)d_in[23];
    const float* oW0 = (const float*)d_in[24]; const float* ob0 = (const float*)d_in[25];
    const float* oW1 = (const float*)d_in[26]; const float* ob1 = (const float*)d_in[27];
    const int* absorber = (const int*)d_in[28];

    char* ws = (char*)d_ws;
    size_t off = 0;
    auto alloc = [&](size_t bytes) -> void* {
        void* p = (void*)(ws + off);
        off += (bytes + 255) & ~(size_t)255;
        return p;
    };

    half_t* wq0 = (half_t*)alloc((size_t)QIN_DIM * HIDDEN_ * 2);
    half_t* wq1 = (half_t*)alloc((size_t)HIDDEN_ * HIDDEN_ * 2);
    half_t* wq2 = (half_t*)alloc((size_t)HIDDEN_ * LATENT_ * 2);
    half_t* wk0 = (half_t*)alloc((size_t)STAT_PAD * HIDDEN_ * 2);
    half_t* wk1 = (half_t*)alloc((size_t)HIDDEN_ * HIDDEN_ * 2);
    half_t* wk2 = (half_t*)alloc((size_t)HIDDEN_ * LATENT_ * 2);
    half_t* wv0 = (half_t*)alloc((size_t)STAT_PAD * HIDDEN_ * 2);
    half_t* wv1 = (half_t*)alloc((size_t)HIDDEN_ * HIDDEN_ * 2);
    half_t* wv2 = (half_t*)alloc((size_t)HIDDEN_ * LATENT_ * 2);
    half_t* wo0 = (half_t*)alloc((size_t)LATENT_ * HIDDEN_ * 2);
    half_t* wo1 = (half_t*)alloc((size_t)HIDDEN_ * LATENT_ * 2);

    half_t* qin  = (half_t*)alloc((size_t)B_ * NE_ * QIN_DIM * 2);
    half_t* stat = (half_t*)alloc((size_t)B_ * N_ * STAT_PAD * 2);
    half_t* t0   = (half_t*)alloc((size_t)B_ * NE_ * HIDDEN_ * 2);
    half_t* t1   = (half_t*)alloc((size_t)B_ * NE_ * HIDDEN_ * 2);
    half_t* qbuf = (half_t*)alloc((size_t)B_ * NE_ * LATENT_ * 2);
    half_t* kbuf = (half_t*)alloc((size_t)B_ * N_ * LATENT_ * 2);
    half_t* vbuf = (half_t*)alloc((size_t)B_ * N_ * LATENT_ * 2);
    half_t* abuf = (half_t*)alloc((size_t)B_ * NE_ * LATENT_ * 2);
    float*  biasArr  = (float*)alloc((size_t)B_ * N_ * 4);
    float*  validArr = (float*)alloc((size_t)B_ * N_ * 4);

    auto conv = [&](const float* src, half_t* dst, int Ksrc, int Kpad, int Ncols) {
        long long total = (long long)Kpad * Ncols;
        int blocks = (int)((total + 255) / 256);
        convert_pad_f16_kernel<<<blocks, 256, 0, stream>>>(src, dst, Ksrc, Kpad, Ncols);
    };
    conv(qW0, wq0, QIN_DIM, QIN_DIM, HIDDEN_);
    conv(qW1, wq1, HIDDEN_, HIDDEN_, HIDDEN_);
    conv(qW2, wq2, HIDDEN_, HIDDEN_, LATENT_);
    conv(kW0, wk0, STAT_RAW, STAT_PAD, HIDDEN_);
    conv(kW1, wk1, HIDDEN_, HIDDEN_, HIDDEN_);
    conv(kW2, wk2, HIDDEN_, HIDDEN_, LATENT_);
    conv(vW0, wv0, STAT_RAW, STAT_PAD, HIDDEN_);
    conv(vW1, wv1, HIDDEN_, HIDDEN_, HIDDEN_);
    conv(vW2, wv2, HIDDEN_, HIDDEN_, LATENT_);
    conv(oW0, wo0, LATENT_, LATENT_, HIDDEN_);
    conv(oW1, wo1, HIDDEN_, HIDDEN_, LATENT_);

    build_static_kernel<<<B_ * N_, 128, 0, stream>>>(h, z, pos, mask, z_emb,
                                                     absorber, stat, biasArr, validArr);
    build_qin_kernel<<<B_ * NE_, 128, 0, stream>>>(h, e_feat, absorber, qin);

    // mode: 0 = SiLU + f16 out, 1 = f16 out, 2 = f32 out
    auto gemm = [&](const half_t* Ap, const half_t* Wp, const float* bp,
                    half_t* c16, float* c32, int M, int Nn, int Kk, int mode) {
        dim3 grid(Nn / 128, M / 128);
        if (mode == 0)
            gemm_f16_kernel<1, true, false><<<grid, 256, 0, stream>>>(Ap, Wp, bp, c16, c32, M, Nn, Kk);
        else if (mode == 1)
            gemm_f16_kernel<0, true, false><<<grid, 256, 0, stream>>>(Ap, Wp, bp, c16, c32, M, Nn, Kk);
        else
            gemm_f16_kernel<0, false, true><<<grid, 256, 0, stream>>>(Ap, Wp, bp, c16, c32, M, Nn, Kk);
    };

    const int MQ = B_ * NE_;   // 16384
    const int MA = B_ * N_;    // 4096

    // q MLP
    gemm(qin, wq0, qb0, t0,   nullptr, MQ, HIDDEN_, QIN_DIM, 0);
    gemm(t0,  wq1, qb1, t1,   nullptr, MQ, HIDDEN_, HIDDEN_, 0);
    gemm(t1,  wq2, qb2, qbuf, nullptr, MQ, LATENT_, HIDDEN_, 1);
    // k MLP
    gemm(stat, wk0, kb0, t0,   nullptr, MA, HIDDEN_, STAT_PAD, 0);
    gemm(t0,   wk1, kb1, t1,   nullptr, MA, HIDDEN_, HIDDEN_, 0);
    gemm(t1,   wk2, kb2, kbuf, nullptr, MA, LATENT_, HIDDEN_, 1);
    // v MLP
    gemm(stat, wv0, vb0, t0,   nullptr, MA, HIDDEN_, STAT_PAD, 0);
    gemm(t0,   wv1, vb1, t1,   nullptr, MA, HIDDEN_, HIDDEN_, 0);
    gemm(t1,   wv2, vb2, vbuf, nullptr, MA, LATENT_, HIDDEN_, 1);

    // fused attention
    attention_kernel<<<dim3(NE_ / 32, NHEADS_, B_), 256, 0, stream>>>(
        qbuf, kbuf, vbuf, biasArr, validArr, abuf);

    // o MLP (final layer writes fp32 directly to d_out)
    gemm(abuf, wo0, ob0, t0, nullptr, MQ, HIDDEN_, LATENT_, 0);
    gemm(t0,   wo1, ob1, nullptr, (float*)d_out, MQ, LATENT_, HIDDEN_, 2);
}
